// SymbolicReasoner_37864431681688
// MI455X (gfx1250) — compile-verified, matrix operand order checked by reading, not run on hardware
//
#include <hip/hip_runtime.h>
#include <hip/hip_bf16.h>
#include <math.h>

// ---------------------------------------------------------------------------
// SymbolicReasoner on MI455X (gfx1250, wave32, WMMA)
// B=4, S=4096, H=1024, STEPS=4, tokens BS=16384
//
// Roofline: ~1.24 TFLOP bf16 WMMA, ~160MB HBM traffic/step -> memory bound.
// Weights (18MB bf16) + activations (32MB bf16) are L2-resident (192MB L2).
// ---------------------------------------------------------------------------

typedef __bf16 bf16_t;
typedef bf16_t v16bf __attribute__((ext_vector_type(16)));
typedef bf16_t v8bf  __attribute__((ext_vector_type(8)));
typedef float  v8f   __attribute__((ext_vector_type(8)));

static constexpr int    H    = 1024;
static constexpr int    SEQ  = 4096;
static constexpr int    BS   = 16384;          // B*S tokens
static constexpr int    HV   = 512;            // H/2 verifier hidden
static constexpr size_t SZ_BIN = 2048ull * 1024ull;   // packed elems, binary op
static constexpr size_t SZ_UN  = 1024ull * 1024ull;   // packed elems, unary op
// bf16-element offsets inside the packed-weights region
static constexpr size_t OFF_AND = 0;
static constexpr size_t OFF_OR  = SZ_BIN;
static constexpr size_t OFF_IMP = 2 * SZ_BIN;
static constexpr size_t OFF_NOT = 3 * SZ_BIN;
static constexpr size_t OFF_FOR = 3 * SZ_BIN + SZ_UN;
static constexpr size_t OFF_EXI = 3 * SZ_BIN + 2 * SZ_UN;
static constexpr size_t OFF_V1  = 3 * SZ_BIN + 3 * SZ_UN;

// --------------------------- helpers ---------------------------------------

// A-fragment (16x32 bf16, ISA 7.12.2): lane<16 holds row M=lane, K={k..k+7, k+16..k+23};
// lane>=16 holds row M=lane-16, K={k+8..k+15, k+24..k+31}. Two 16B loads.
__device__ inline v16bf load_a_frag(const bf16_t* __restrict__ p) {
  v8bf lo = *(const v8bf*)(p);
  v8bf hi = *(const v8bf*)(p + 16);
  v16bf a;
#pragma unroll
  for (int i = 0; i < 8; ++i) { a[i] = lo[i]; a[i + 8] = hi[i]; }
  return a;
}

// --------------------------- pack kernels ----------------------------------

// Pack fp32 weight W[K x N] (row-major, N = fan_out) into bf16 B-fragments:
// dst[((nt*nKC + kc)*512) + lane*16 + e] = W[k, n]
//   k = kc*32 + (lane>>4)*8 + (e<8 ? e : e+8),  n = nt*16 + (lane&15)
// so the GEMM reads one contiguous 32B v16bf per lane per WMMA.
__global__ void pack_w_kernel(const float* __restrict__ W, bf16_t* __restrict__ dst,
                              int K, int N) {
  int tid  = blockIdx.x * 256 + threadIdx.x;
  int e    = tid & 15;
  int lane = (tid >> 4) & 31;
  int frag = tid >> 9;
  int nKC  = K >> 5;
  int kc   = frag % nKC;
  int nt   = frag / nKC;
  int k = kc * 32 + ((lane >> 4) << 3) + (e < 8 ? e : e + 8);
  int n = (nt << 4) + (lane & 15);
  dst[(size_t)frag * 512 + (size_t)lane * 16 + e] = (bf16_t)W[(size_t)k * N + n];
}

__global__ void cvt_bf16_kernel(const float* __restrict__ src, bf16_t* __restrict__ dst,
                                int n) {
  int i = blockIdx.x * 256 + threadIdx.x;
  if (i < n) dst[i] = (bf16_t)src[i];
}

// --------------------------- coefficient kernel -----------------------------
// Per token: is_operator = softmax(structure_logits)[0]; op_probs = softmax(emb0@Wsel+bsel).
// Blend order [and(0), or(1), implies(3), not(2), forall(4), exists(5)] collapses to
//   emb' = cEmb*emb + sum_i c_i*(y_i)   with c_i = w_i * prod_{j>i}(1-w_j), cEmb = prod(1-w_j).
// These are step-invariant, so compute once. One wave per token.
__global__ void coef_kernel(const float* __restrict__ emb0, const float* __restrict__ slog,
                            const float* __restrict__ Wsel, const float* __restrict__ bsel,
                            float* __restrict__ coef) {
  int wave = threadIdx.x >> 5, lane = threadIdx.x & 31;
  int t = blockIdx.x * 8 + wave;
  const float* row = emb0 + (size_t)t * H;
  float s[6] = {0.f, 0.f, 0.f, 0.f, 0.f, 0.f};
  for (int k = lane; k < H; k += 32) {
    float x = row[k];
#pragma unroll
    for (int o = 0; o < 6; ++o) s[o] += x * Wsel[k * 6 + o];
  }
#pragma unroll
  for (int o = 0; o < 6; ++o)
    for (int off = 16; off; off >>= 1) s[o] += __shfl_xor(s[o], off, 32);
  if (lane == 0) {
    float l[6], p[6];
    float mx = -1e30f;
#pragma unroll
    for (int o = 0; o < 6; ++o) { l[o] = s[o] + bsel[o]; mx = fmaxf(mx, l[o]); }
    float den = 0.f;
#pragma unroll
    for (int o = 0; o < 6; ++o) { p[o] = expf(l[o] - mx); den += p[o]; }
#pragma unroll
    for (int o = 0; o < 6; ++o) p[o] /= den;
    float a0 = slog[(size_t)t * 3 + 0], a1 = slog[(size_t)t * 3 + 1], a2 = slog[(size_t)t * 3 + 2];
    float am = fmaxf(a0, fmaxf(a1, a2));
    float e0 = expf(a0 - am), e1 = expf(a1 - am), e2 = expf(a2 - am);
    float isop = e0 / (e0 + e1 + e2);
    // blend order: and(0), or(1), implies(3), not(2), forall(4), exists(5)
    float w[6] = { isop * p[0], isop * p[1], isop * p[3],
                   isop * p[2], isop * p[4], isop * p[5] };
    float cc[6], prod = 1.f;
#pragma unroll
    for (int i = 5; i >= 0; --i) { cc[i] = w[i] * prod; prod *= (1.f - w[i]); }
    float* c = coef + (size_t)t * 8;
    c[0] = prod;
#pragma unroll
    for (int i = 0; i < 6; ++i) c[1 + i] = cc[i];
    c[7] = 0.f;
  }
}

// --------------------------- fused step GEMM --------------------------------
// grid = (H/16 = 64 n-tiles, BS/128 = 128 m-supertiles), block = 256 (8 waves).
// Each wave owns one 16x16 output tile with 6 f32 accumulators (one per op);
// each A-fragment feeds 6 WMMAs (3 for the rolled-row K-sweep).
__global__ void __launch_bounds__(256)
step_gemm_kernel(const bf16_t* __restrict__ Abf, const bf16_t* __restrict__ Bpack,
                 const float* __restrict__ coef,
                 const float* __restrict__ b0, const float* __restrict__ b1,
                 const float* __restrict__ b2, const float* __restrict__ b3,
                 const float* __restrict__ b4, const float* __restrict__ b5,
                 bf16_t* __restrict__ outBf, float* __restrict__ outF32) {
  __shared__ float sCoef[128 * 8];
  const int tid  = threadIdx.x;
  const int wave = tid >> 5, lane = tid & 31;
  const int mBlock = blockIdx.y * 128;
  const int nTile  = blockIdx.x;

  for (int i = tid; i < 128 * 8; i += 256) sCoef[i] = coef[(size_t)mBlock * 8 + i];
  __syncthreads();

  const int mBase   = mBlock + wave * 16;
  const int rowCur  = mBase + (lane & 15);
  const int rowPrev = ((rowCur & (SEQ - 1)) == 0) ? rowCur + SEQ - 1 : rowCur - 1;
  const int kOff    = (lane >> 4) << 3;
  const int bLane   = lane * 16;

  const bf16_t* aCur  = Abf + (size_t)rowCur * H;
  const bf16_t* aPrev = Abf + (size_t)rowPrev * H;
  const bf16_t* bp[6] = {
    Bpack + OFF_AND + (size_t)nTile * (64 * 512),
    Bpack + OFF_OR  + (size_t)nTile * (64 * 512),
    Bpack + OFF_IMP + (size_t)nTile * (64 * 512),
    Bpack + OFF_NOT + (size_t)nTile * (32 * 512),
    Bpack + OFF_FOR + (size_t)nTile * (32 * 512),
    Bpack + OFF_EXI + (size_t)nTile * (32 * 512)
  };

  v8f acc[6] = {};
  // K-sweep 1: current rows feed all 6 ops (binary top half + unary full)
  for (int kc = 0; kc < 32; ++kc) {
    v16bf a = load_a_frag(aCur + kc * 32 + kOff);
#pragma unroll
    for (int o = 0; o < 6; ++o) {
      v16bf b = *(const v16bf*)(bp[o] + (size_t)kc * 512 + bLane);
      acc[o] = __builtin_amdgcn_wmma_f32_16x16x32_bf16(false, a, false, b,
                                                       (short)0, acc[o], false, false);
    }
  }
  // K-sweep 2: rolled rows feed binary ops' bottom halves
  for (int kc = 0; kc < 32; ++kc) {
    v16bf a = load_a_frag(aPrev + kc * 32 + kOff);
#pragma unroll
    for (int o = 0; o < 3; ++o) {
      v16bf b = *(const v16bf*)(bp[o] + (size_t)(32 + kc) * 512 + bLane);
      acc[o] = __builtin_amdgcn_wmma_f32_16x16x32_bf16(false, a, false, b,
                                                       (short)0, acc[o], false, false);
    }
  }

  // Epilogue: emb' = cEmb*emb + sum_o c_o*(acc_o + bias_o)
  const int col = nTile * 16 + (lane & 15);
  const float bias[6] = { b0[col], b1[col], b2[col], b3[col], b4[col], b5[col] };
#pragma unroll
  for (int r = 0; r < 8; ++r) {
    int row = mBase + r + ((lane >> 4) << 3);          // C-layout: M = r + 8*(lane>=16)
    const float* c = &sCoef[(row - mBlock) * 8];
    float v = c[0] * (float)Abf[(size_t)row * H + col];
#pragma unroll
    for (int o = 0; o < 6; ++o) v += c[1 + o] * (acc[o][r] + bias[o]);
    outBf[(size_t)row * H + col] = (bf16_t)v;
    if (outF32) outF32[(size_t)row * H + col] = v;
  }
}

// --------------------------- verifier head ----------------------------------
// h = gelu(emb @ W_v1 + b_v1): WMMA GEMM M=16384, N=512, K=1024
__global__ void __launch_bounds__(256)
v1_kernel(const bf16_t* __restrict__ Abf, const bf16_t* __restrict__ Bv1,
          const float* __restrict__ bv1, bf16_t* __restrict__ hOut) {
  const int tid  = threadIdx.x;
  const int wave = tid >> 5, lane = tid & 31;
  const int mBase = blockIdx.y * 128 + wave * 16;
  const int nTile = blockIdx.x;
  const int rowCur = mBase + (lane & 15);
  const int kOff   = (lane >> 4) << 3;
  const bf16_t* aCur = Abf + (size_t)rowCur * H;
  const bf16_t* bp   = Bv1 + (size_t)nTile * (32 * 512);
  const int bLane = lane * 16;

  v8f acc = {};
  for (int kc = 0; kc < 32; ++kc) {
    v16bf a = load_a_frag(aCur + kc * 32 + kOff);
    v16bf b = *(const v16bf*)(bp + (size_t)kc * 512 + bLane);
    acc = __builtin_amdgcn_wmma_f32_16x16x32_bf16(false, a, false, b,
                                                  (short)0, acc, false, false);
  }
  const int col = nTile * 16 + (lane & 15);
  const float bias = bv1[col];
#pragma unroll
  for (int r = 0; r < 8; ++r) {
    int row = mBase + r + ((lane >> 4) << 3);
    float x = acc[r] + bias;
    float g = 0.5f * x * (1.0f + erff(x * 0.70710678118654752f));   // exact gelu
    hOut[(size_t)row * HV + col] = (bf16_t)g;
  }
}

// scores = sigmoid(h @ W_v2 + b_v2): one wave per token, shuffle reduction
__global__ void v2_kernel(const bf16_t* __restrict__ h, const float* __restrict__ Wv2,
                          const float* __restrict__ bv2, float* __restrict__ out) {
  int wave = threadIdx.x >> 5, lane = threadIdx.x & 31;
  int t = blockIdx.x * 8 + wave;
  const bf16_t* row = h + (size_t)t * HV;
  float s = 0.f;
  for (int k = lane; k < HV; k += 32) s += (float)row[k] * Wv2[k];
  for (int off = 16; off; off >>= 1) s += __shfl_xor(s, off, 32);
  if (lane == 0) out[t] = 1.0f / (1.0f + expf(-(s + bv2[0])));
}

// --------------------------- host launcher ----------------------------------

extern "C" void kernel_launch(void* const* d_in, const int* in_sizes, int n_in,
                              void* d_out, int out_size, void* d_ws, size_t ws_size,
                              hipStream_t stream) {
  (void)in_sizes; (void)n_in; (void)out_size; (void)ws_size;
  // input order: 0 emb, 1 slog, 2/3 W,b_and, 4/5 or, 6/7 not, 8/9 implies,
  //              10/11 forall, 12/13 exists, 14/15 sel, 16/17 v1, 18/19 v2
  const float* emb0 = (const float*)d_in[0];
  const float* slog = (const float*)d_in[1];

  char* ws = (char*)d_ws;
  bf16_t* Bpack = (bf16_t*)(ws);                     // 19.0 MiB packed bf16 weights
  bf16_t* bfA   = (bf16_t*)(ws + 20971520);          // 32 MiB bf16 emb ping
  bf16_t* bfB   = (bf16_t*)(ws + 54525952);          // 32 MiB bf16 emb pong
  float*  coef  = (float*) (ws + 88080384);          // 512 KiB per-token coefs
  bf16_t* hbuf  = (bf16_t*)(ws + 88604672);          // 16 MiB verifier hidden
  // total ws use ~100.5 MiB

  // 1) pack weights -> bf16 WMMA B-fragment layout (deterministic, every call)
  auto packW = [&](int idx, size_t off, int K, int N) {
    pack_w_kernel<<<(K * N) / 256, 256, 0, stream>>>((const float*)d_in[idx],
                                                     Bpack + off, K, N);
  };
  packW(2,  OFF_AND, 2048, 1024);
  packW(4,  OFF_OR,  2048, 1024);
  packW(8,  OFF_IMP, 2048, 1024);
  packW(6,  OFF_NOT, 1024, 1024);
  packW(10, OFF_FOR, 1024, 1024);
  packW(12, OFF_EXI, 1024, 1024);
  packW(16, OFF_V1,  1024, 512);

  // 2) bf16 copy of the input embeddings (input is never mutated)
  cvt_bf16_kernel<<<(BS * H) / 256, 256, 0, stream>>>(emb0, bfA, BS * H);

  // 3) step-invariant per-token blend coefficients
  coef_kernel<<<BS / 8, 256, 0, stream>>>(emb0, slog, (const float*)d_in[14],
                                          (const float*)d_in[15], coef);

  // 4) four fused reasoning steps (ping-pong bf16; last step also writes fp32 d_out)
  const float* bAnd = (const float*)d_in[3];
  const float* bOr  = (const float*)d_in[5];
  const float* bImp = (const float*)d_in[9];
  const float* bNot = (const float*)d_in[7];
  const float* bFor = (const float*)d_in[11];
  const float* bExi = (const float*)d_in[13];
  dim3 g(H / 16, BS / 128);
  step_gemm_kernel<<<g, 256, 0, stream>>>(bfA, Bpack, coef, bAnd, bOr, bImp, bNot, bFor, bExi,
                                          bfB, (float*)nullptr);
  step_gemm_kernel<<<g, 256, 0, stream>>>(bfB, Bpack, coef, bAnd, bOr, bImp, bNot, bFor, bExi,
                                          bfA, (float*)nullptr);
  step_gemm_kernel<<<g, 256, 0, stream>>>(bfA, Bpack, coef, bAnd, bOr, bImp, bNot, bFor, bExi,
                                          bfB, (float*)nullptr);
  step_gemm_kernel<<<g, 256, 0, stream>>>(bfB, Bpack, coef, bAnd, bOr, bImp, bNot, bFor, bExi,
                                          bfA, (float*)d_out);   // final emb -> d_out[0:BS*H]

  // 5) verifier head
  v1_kernel<<<dim3(HV / 16, BS / 128), 256, 0, stream>>>(bfA, Bpack + OFF_V1,
                                                         (const float*)d_in[17], hbuf);
  v2_kernel<<<BS / 8, 256, 0, stream>>>(hbuf, (const float*)d_in[18],
                                        (const float*)d_in[19],
                                        (float*)d_out + (size_t)BS * H);
}
